// BiMambaBlockAdaLN_65163243815252
// MI455X (gfx1250) — compile-verified, matrix operand order checked by reading, not run on hardware
//
#include <hip/hip_runtime.h>
#include <hip/hip_bf16.h>
#include <math.h>

// ---------------- problem constants (from reference) ----------------
#define DIMC   512
#define SEQL   2048
#define BSZ    4
#define DI     1024          // D_INNER
#define DSTATE 16
#define DTRANK 32
#define MROWS  (BSZ * SEQL)  // 8192 token rows

typedef __attribute__((ext_vector_type(16))) _Float16 v16h;
typedef __attribute__((ext_vector_type(8)))  _Float16 v8h;
typedef __attribute__((ext_vector_type(8)))  float    v8f;

__device__ __forceinline__ float siluf(float x)     { return x / (1.f + __expf(-x)); }
__device__ __forceinline__ float softplusf(float x) { return x > 20.f ? x : log1pf(__expf(x)); }
__device__ __forceinline__ float geluf(float x)     { return 0.5f * x * (1.f + erff(x * 0.70710678118654752440f)); }

// -------------------------------------------------------------------
// WMMA fragment load for a K-contiguous (row-major, row length ldk) f16
// matrix. Per CDNA5 ISA 16-bit A/B layout (16x32 / 32x16):
//   lanes 0-15  : row = lane,    K = [0..7]  in v0-3, [16..23] in v4-7
//   lanes 16-31 : row = lane-16, K = [8..15] in v0-3, [24..31] in v4-7
// -------------------------------------------------------------------
__device__ __forceinline__ v16h load_frag_k(const _Float16* __restrict__ rowbase, int ldk) {
    const int lane = threadIdx.x & 31;
    const int r    = lane & 15;
    const int hi   = lane >> 4;
    const _Float16* p = rowbase + (size_t)r * ldk + (hi << 3);
    v8h a = *(const v8h*)(p);
    v8h b = *(const v8h*)(p + 16);
    v16h f;
#pragma unroll
    for (int i = 0; i < 8; ++i) { f[i] = a[i]; f[i + 8] = b[i]; }
    return f;
}

// ---------------- GEMM: C(MxN) = A(MxK) * W(NxK)^T, f16 in, f32 acc ----------------
// Block = 128 threads (4 waves). Block tile 128x64; each wave owns a 32x64
// strip (2 M-tiles x 4 N-tiles = 8 WMMAs per K-step). K is a template
// parameter so the K-loop is fully unrolled straight-line SSA: the scheduler
// hoists fragment loads across steps and accumulators stay strictly in-place
// (no rotation moves, no accumulator copies, no hot-path branches).
enum {
    EPI_PLAIN = 0,   // C f32
    EPI_XZ,          // N==2048: cols<1024 -> xc f16; cols>=1024 -> silu -> sz f16
    EPI_DBL,         // C f32 (dbl, ld 64) + first DTRANK cols as f16 (dtr, ld 32)
    EPI_DT,          // softplus(val + bias[n]) -> f16 (ld N)
    EPI_OUT_ADD,     // C = res + val        (fwd out-proj: res = block input x)
    EPI_OUT_ACC,     // C += val             (bwd out-proj accumulate)
    EPI_GELU,        // gelu(val + bias[n]) -> f16 (ld N)
    EPI_FFN2         // C = val + bias[n] + res   (final output)
};

template <int EPI, int K>
__global__ __launch_bounds__(128) void gemm_wmma_kernel(
    const _Float16* __restrict__ A, const _Float16* __restrict__ Bw,
    float* __restrict__ C, _Float16* __restrict__ O16a, _Float16* __restrict__ O16b,
    const float* __restrict__ bias, const float* __restrict__ res,
    int M, int N)
{
    const int wave  = threadIdx.x >> 5;
    const int lane  = threadIdx.x & 31;
    const int mbase = blockIdx.x * 128 + wave * 32;  // each wave: 2 x 16 rows
    const int nbase = blockIdx.y * 64;               // each wave: 4 x 16 cols

    v8f acc[2][4] = {};

    const _Float16* Ar[2] = { A + (size_t)mbase * K, A + (size_t)(mbase + 16) * K };
    const _Float16* Br[4] = { Bw + (size_t)(nbase +  0) * K, Bw + (size_t)(nbase + 16) * K,
                              Bw + (size_t)(nbase + 32) * K, Bw + (size_t)(nbase + 48) * K };

#pragma unroll
    for (int kb = 0; kb < K; kb += 32) {
        v16h aS[2], bS[4];
        aS[0] = load_frag_k(Ar[0] + kb, K);
        aS[1] = load_frag_k(Ar[1] + kb, K);
        bS[0] = load_frag_k(Br[0] + kb, K);
        bS[1] = load_frag_k(Br[1] + kb, K);
        bS[2] = load_frag_k(Br[2] + kb, K);
        bS[3] = load_frag_k(Br[3] + kb, K);
#pragma unroll
        for (int m = 0; m < 2; ++m)
#pragma unroll
            for (int t = 0; t < 4; ++t)
                acc[m][t] = __builtin_amdgcn_wmma_f32_16x16x32_f16(
                    false, aS[m], false, bS[t], (short)0, acc[m][t], false, false);
    }

    const int hi  = lane >> 4;
    const int nc0 = lane & 15;
#pragma unroll
    for (int m = 0; m < 2; ++m) {
#pragma unroll
        for (int t = 0; t < 4; ++t) {
#pragma unroll
            for (int v = 0; v < 8; ++v) {
                const int   mr  = mbase + m * 16 + hi * 8 + v;
                const int   nc  = nbase + t * 16 + nc0;
                const float val = acc[m][t][v];
                if constexpr (EPI == EPI_PLAIN) {
                    C[(size_t)mr * N + nc] = val;
                } else if constexpr (EPI == EPI_XZ) {
                    if (nc < DI) O16a[(size_t)mr * DI + nc]        = (_Float16)val;
                    else         O16b[(size_t)mr * DI + (nc - DI)] = (_Float16)siluf(val);
                } else if constexpr (EPI == EPI_DBL) {
                    C[(size_t)mr * 64 + nc] = val;
                    if (nc < DTRANK) O16a[(size_t)mr * DTRANK + nc] = (_Float16)val;
                } else if constexpr (EPI == EPI_DT) {
                    O16a[(size_t)mr * N + nc] = (_Float16)softplusf(val + bias[nc]);
                } else if constexpr (EPI == EPI_OUT_ADD) {
                    C[(size_t)mr * N + nc] = res[(size_t)mr * N + nc] + val;
                } else if constexpr (EPI == EPI_OUT_ACC) {
                    C[(size_t)mr * N + nc] += val;
                } else if constexpr (EPI == EPI_GELU) {
                    O16a[(size_t)mr * N + nc] = (_Float16)geluf(val + bias[nc]);
                } else if constexpr (EPI == EPI_FFN2) {
                    C[(size_t)mr * N + nc] = val + bias[nc] + res[(size_t)mr * N + nc];
                }
            }
        }
    }
}

// ---------------- f32 -> f16 weight cast ----------------
__global__ void cast_f16_kernel(const float* __restrict__ in, _Float16* __restrict__ out, int n) {
    int i = blockIdx.x * 256 + threadIdx.x;
    if (i < n) out[i] = (_Float16)in[i];
}

// ---------------- AdaLN: mod = silu(cond) @ ada_W^T + ada_b ----------------
__global__ __launch_bounds__(256) void ada_kernel(
    const float* __restrict__ cond, const float* __restrict__ W,
    const float* __restrict__ bb, float* __restrict__ mod)
{
    __shared__ float sc[DIMC];
    const int b   = blockIdx.y;
    const int tid = threadIdx.x;
    sc[tid]       = siluf(cond[b * DIMC + tid]);
    sc[tid + 256] = siluf(cond[b * DIMC + tid + 256]);
    __syncthreads();
    const int o = blockIdx.x * 256 + tid;
    float acc = bb[o];
    for (int k = 0; k < DIMC; ++k) acc += sc[k] * W[(size_t)o * DIMC + k];
    mod[(size_t)b * 4 * DIMC + o] = acc;
}

// ---------------- LayerNorm + AdaLN modulation -> f16 ----------------
__global__ __launch_bounds__(256) void ln_mod_kernel(
    const float* __restrict__ X, const float* __restrict__ mod,
    _Float16* __restrict__ H, int shift_off, int scale_off)
{
    __shared__ float red[256];
    const int row = blockIdx.x;
    const int b   = row / SEQL;
    const int tid = threadIdx.x;
    const float* xr = X + (size_t)row * DIMC;
    float v0 = xr[tid], v1 = xr[tid + 256];

    red[tid] = v0 + v1; __syncthreads();
    for (int s = 128; s > 0; s >>= 1) { if (tid < s) red[tid] += red[tid + s]; __syncthreads(); }
    const float mu = red[0] * (1.f / DIMC);
    __syncthreads();
    const float d0 = v0 - mu, d1 = v1 - mu;
    red[tid] = d0 * d0 + d1 * d1; __syncthreads();
    for (int s = 128; s > 0; s >>= 1) { if (tid < s) red[tid] += red[tid + s]; __syncthreads(); }
    const float rs = rsqrtf(red[0] * (1.f / DIMC) + 1e-6f);

    const float* mb = mod + (size_t)b * (4 * DIMC);
    H[(size_t)row * DIMC + tid] =
        (_Float16)(d0 * rs * (1.f + mb[scale_off + tid]) + mb[shift_off + tid]);
    H[(size_t)row * DIMC + tid + 256] =
        (_Float16)(d1 * rs * (1.f + mb[scale_off + tid + 256]) + mb[shift_off + tid + 256]);
}

// ---------------- depthwise causal conv (4 taps) + SiLU ----------------
// dir==0: taps t-3..t (forward causal); dir==1: taps t..t+3 (time-reversed
// branch kept in physical time: flip(conv(flip(x))) == future-looking taps).
__global__ void conv_silu_kernel(
    const _Float16* __restrict__ xc, const float* __restrict__ w,
    const float* __restrict__ cb, _Float16* __restrict__ u, int dir)
{
    const int idx = blockIdx.x * 256 + threadIdx.x;
    if (idx >= MROWS * DI) return;
    const int d = idx % DI;
    const int t = (idx / DI) % SEQL;
    const int b = idx / (DI * SEQL);
    float acc = cb[d];
#pragma unroll
    for (int j = 0; j < 4; ++j) {
        const int tt = (dir == 0) ? (t - 3 + j) : (t + 3 - j);
        if (tt >= 0 && tt < SEQL)
            acc += w[d * 4 + j] * (float)xc[((size_t)(b * SEQL + tt)) * DI + d];
    }
    u[idx] = (_Float16)siluf(acc);
}

// ---------------- selective scan: one channel per lane, 16 states in regs ----------------
// Latency-bound dependence chain over SEQL steps; prefetch rows PF steps ahead
// (global_prefetch_b8) so per-step loads hit cache.
__global__ __launch_bounds__(256) void scan_kernel(
    const _Float16* __restrict__ dt16, const _Float16* __restrict__ u16,
    const float* __restrict__ dbl, const float* __restrict__ Alog,
    const float* __restrict__ Dv, const _Float16* __restrict__ sz16,
    _Float16* __restrict__ ya, int dir)
{
    const int d = blockIdx.x * 256 + threadIdx.x;
    const int b = blockIdx.y;
    float Am[DSTATE], h[DSTATE];
#pragma unroll
    for (int n = 0; n < DSTATE; ++n) { Am[n] = -__expf(Alog[d * DSTATE + n]); h[n] = 0.f; }
    const float Dd = Dv[d];
    const int PF = 8;

    for (int s = 0; s < SEQL; ++s) {
        if (s + PF < SEQL) {
            const int    tp = (dir == 0) ? (s + PF) : (SEQL - 1 - (s + PF));
            const size_t rp = (size_t)b * SEQL + tp;
            __builtin_prefetch(dt16 + rp * DI + d, 0, 1);
            __builtin_prefetch(u16  + rp * DI + d, 0, 1);
            __builtin_prefetch(sz16 + rp * DI + d, 0, 1);
            __builtin_prefetch(dbl  + rp * 64 + 32, 0, 1);
        }
        const int    t   = (dir == 0) ? s : (SEQL - 1 - s);
        const size_t row = (size_t)b * SEQL + t;
        const float dtv = (float)dt16[row * DI + d];
        const float uv  = (float)u16[row * DI + d];
        const float dtu = dtv * uv;
        const float* bc = dbl + row * 64;   // wave-uniform address -> scalar loads
        float acc = 0.f;
#pragma unroll
        for (int n = 0; n < DSTATE; ++n) {
            const float dA = __expf(dtv * Am[n]);
            h[n] = dA * h[n] + dtu * bc[32 + n];
            acc += h[n] * bc[48 + n];
        }
        const float y = acc + Dd * uv;
        ya[row * DI + d] = (_Float16)(y * (float)sz16[row * DI + d]);
    }
}

// ===================================================================
extern "C" void kernel_launch(void* const* d_in, const int* in_sizes, int n_in,
                              void* d_out, int out_size, void* d_ws, size_t ws_size,
                              hipStream_t stream) {
    (void)in_sizes; (void)n_in; (void)out_size; (void)ws_size;

    const float* x      = (const float*)d_in[0];
    const float* cond   = (const float*)d_in[1];
    const float* ada_W  = (const float*)d_in[2];
    const float* ada_b  = (const float*)d_in[3];
    const float* ffn_W1 = (const float*)d_in[4];
    const float* ffn_b1 = (const float*)d_in[5];
    const float* ffn_W2 = (const float*)d_in[6];
    const float* ffn_b2 = (const float*)d_in[7];

    struct MP { const float *Win, *convw, *convb, *Wx, *Wdt, *bdt, *Alog, *D, *Wout; };
    MP mp[2];
    for (int dir = 0; dir < 2; ++dir) {
        const int o = 8 + dir * 9;
        mp[dir].Win   = (const float*)d_in[o + 0];
        mp[dir].convw = (const float*)d_in[o + 1];
        mp[dir].convb = (const float*)d_in[o + 2];
        mp[dir].Wx    = (const float*)d_in[o + 3];
        mp[dir].Wdt   = (const float*)d_in[o + 4];
        mp[dir].bdt   = (const float*)d_in[o + 5];
        mp[dir].Alog  = (const float*)d_in[o + 6];
        mp[dir].D     = (const float*)d_in[o + 7];
        mp[dir].Wout  = (const float*)d_in[o + 8];
    }
    float* out = (float*)d_out;

    // ----- workspace carve-out -----
    char*  ws  = (char*)d_ws;
    size_t cur = 0;
    auto alloc = [&](size_t bytes) -> void* {
        void* p = ws + cur;
        cur = (cur + bytes + 255) & ~(size_t)255;
        return p;
    };

    float*     mod   = (float*)    alloc(sizeof(float) * BSZ * 4 * DIMC);
    _Float16*  h16   = (_Float16*) alloc(sizeof(_Float16) * (size_t)MROWS * DIMC);
    float*     macc  = (float*)    alloc(sizeof(float) * (size_t)MROWS * DIMC);
    _Float16*  g16   = (_Float16*) alloc(sizeof(_Float16) * (size_t)MROWS * DIMC);
    _Float16*  g1_16 = (_Float16*) alloc(sizeof(_Float16) * (size_t)MROWS * DI);

    _Float16 *win16[2], *wx16[2], *wdt16[2], *wout16[2];
    for (int dir = 0; dir < 2; ++dir) {
        win16[dir]  = (_Float16*)alloc(sizeof(_Float16) * 2048 * 512);
        wx16[dir]   = (_Float16*)alloc(sizeof(_Float16) * 64 * 1024);
        wdt16[dir]  = (_Float16*)alloc(sizeof(_Float16) * 1024 * 32);
        wout16[dir] = (_Float16*)alloc(sizeof(_Float16) * 512 * 1024);
    }
    _Float16* w1_16 = (_Float16*)alloc(sizeof(_Float16) * 1024 * 512);
    _Float16* w2_16 = (_Float16*)alloc(sizeof(_Float16) * 512 * 1024);

    _Float16 *xc16[2], *sz16[2], *u16[2], *dtr16[2], *dt16[2], *ya16[2];
    float* dbl[2];
    for (int dir = 0; dir < 2; ++dir) {
        xc16[dir]  = (_Float16*)alloc(sizeof(_Float16) * (size_t)MROWS * DI);
        sz16[dir]  = (_Float16*)alloc(sizeof(_Float16) * (size_t)MROWS * DI);
        u16[dir]   = (_Float16*)alloc(sizeof(_Float16) * (size_t)MROWS * DI);
        dtr16[dir] = (_Float16*)alloc(sizeof(_Float16) * (size_t)MROWS * DTRANK);
        dt16[dir]  = (_Float16*)alloc(sizeof(_Float16) * (size_t)MROWS * DI);
        ya16[dir]  = (_Float16*)alloc(sizeof(_Float16) * (size_t)MROWS * DI);
        dbl[dir]   = (float*)   alloc(sizeof(float) * (size_t)MROWS * 64);
    }

    auto cast = [&](const float* src, _Float16* dst, int n) {
        cast_f16_kernel<<<(n + 255) / 256, 256, 0, stream>>>(src, dst, n);
    };
    for (int dir = 0; dir < 2; ++dir) {
        cast(mp[dir].Win,  win16[dir],  2048 * 512);
        cast(mp[dir].Wx,   wx16[dir],   64 * 1024);
        cast(mp[dir].Wdt,  wdt16[dir],  1024 * 32);
        cast(mp[dir].Wout, wout16[dir], 512 * 1024);
    }
    cast(ffn_W1, w1_16, 1024 * 512);
    cast(ffn_W2, w2_16, 512 * 1024);

    // AdaLN modulation
    ada_kernel<<<dim3(2048 / 256, BSZ), 256, 0, stream>>>(cond, ada_W, ada_b, mod);

    // h = LN(x) * (1+scale_msa) + shift_msa   (shift@0, scale@512)
    ln_mod_kernel<<<MROWS, 256, 0, stream>>>(x, mod, h16, 0, 512);

    const dim3 gblk(128);
    for (int dir = 0; dir < 2; ++dir) {
        // xz = h @ Win^T : split to xc (f16) and silu(z) (f16)
        gemm_wmma_kernel<EPI_XZ, 512><<<dim3(MROWS / 128, 2048 / 64), gblk, 0, stream>>>(
            h16, win16[dir], nullptr, xc16[dir], sz16[dir], nullptr, nullptr,
            MROWS, 2048);
        // u = silu(depthwise_conv(xc) + b)
        conv_silu_kernel<<<(MROWS * DI) / 256, 256, 0, stream>>>(
            xc16[dir], mp[dir].convw, mp[dir].convb, u16[dir], dir);
        // dbl = u @ Wx^T  (f32, + dt_r slice as f16)
        gemm_wmma_kernel<EPI_DBL, 1024><<<dim3(MROWS / 128, 64 / 64), gblk, 0, stream>>>(
            u16[dir], wx16[dir], dbl[dir], dtr16[dir], nullptr, nullptr, nullptr,
            MROWS, 64);
        // dt = softplus(dt_r @ Wdt^T + bdt)
        gemm_wmma_kernel<EPI_DT, DTRANK><<<dim3(MROWS / 128, 1024 / 64), gblk, 0, stream>>>(
            dtr16[dir], wdt16[dir], nullptr, dt16[dir], nullptr, mp[dir].bdt, nullptr,
            MROWS, 1024);
        // selective scan -> ya = y * silu(z)
        scan_kernel<<<dim3(DI / 256, BSZ), 256, 0, stream>>>(
            dt16[dir], u16[dir], dbl[dir], mp[dir].Alog, mp[dir].D, sz16[dir],
            ya16[dir], dir);
        // out-proj, accumulate into macc (fwd adds residual x, bwd accumulates)
        if (dir == 0)
            gemm_wmma_kernel<EPI_OUT_ADD, 1024><<<dim3(MROWS / 128, 512 / 64), gblk, 0, stream>>>(
                ya16[dir], wout16[dir], macc, nullptr, nullptr, nullptr, x,
                MROWS, 512);
        else
            gemm_wmma_kernel<EPI_OUT_ACC, 1024><<<dim3(MROWS / 128, 512 / 64), gblk, 0, stream>>>(
                ya16[dir], wout16[dir], macc, nullptr, nullptr, nullptr, nullptr,
                MROWS, 512);
    }

    // g = LN(macc) * (1+scale_mlp) + shift_mlp   (shift@1024, scale@1536)
    ln_mod_kernel<<<MROWS, 256, 0, stream>>>(macc, mod, g16, 1024, 1536);

    // g1 = gelu(g @ W1^T + b1)
    gemm_wmma_kernel<EPI_GELU, 512><<<dim3(MROWS / 128, 1024 / 64), gblk, 0, stream>>>(
        g16, w1_16, nullptr, g1_16, nullptr, ffn_b1, nullptr, MROWS, 1024);
    // out = g1 @ W2^T + b2 + macc
    gemm_wmma_kernel<EPI_FFN2, 1024><<<dim3(MROWS / 128, 512 / 64), gblk, 0, stream>>>(
        g1_16, w2_16, out, nullptr, nullptr, ffn_b2, macc, MROWS, 512);
}
